// DifferentialVisionTransformer_59167469469815
// MI455X (gfx1250) — compile-verified, matrix operand order checked by reading, not run on hardware
//
#include <hip/hip_runtime.h>
#include <hip/hip_bf16.h>
#include <math.h>

typedef __attribute__((ext_vector_type(16))) _Float16 v16h;
typedef __attribute__((ext_vector_type(8)))  _Float16 v8h;
typedef __attribute__((ext_vector_type(8)))  float    v8f;

#define B_    16
#define NTOK  197
#define E_    768
#define H_    12
#define HD_   64
#define D_    6
#define MLP_  3072
#define NC_   1000
#define ROWS  (B_*NTOK)    // 3152
#define PROWS (B_*196)     // 3136
#define EPS_  1e-5f

__device__ inline int imin(int a, int b) { return a < b ? a : b; }

// ---------------------------------------------------------------------------
// WMMA helpers (CDNA5 wave32, v_wmma_f32_16x16x32_f16)
// A fragment (16x32 f16): lane<16 -> row=lane, K=0..7 then 16..23
//                         lane>=16 -> row=lane-16, K=8..15 then 24..31
// LDS rows stride 40 halfs (80B) so both 8-half chunks stay 16B aligned.
// ---------------------------------------------------------------------------
__device__ inline v16h ld_frag(const _Float16* row, int hi) {
  union { v16h v; v8h h[2]; } u;
  u.h[0] = *(const v8h*)(row + hi * 8);
  u.h[1] = *(const v8h*)(row + hi * 8 + 16);
  return u.v;
}
__device__ inline v8f wmma_step(v16h a, v16h b, v8f c) {
  return __builtin_amdgcn_wmma_f32_16x16x32_f16(false, a, false, b, (short)0, c,
                                                false, false);
}

// ---------------------------------------------------------------------------
// GEMM: C[M,N] = act(A[M,K] * Bt^T + bias) (+ res). A f16 [M,K], Bt f16 [N,K].
// K must be a multiple of 32. 64(M) x 128(N) block tile; 8 waves, each wave
// computes four 16x16 subtiles (one A frag feeds 4 WMMAs). LDS ping-pong with
// UNCONDITIONAL register prefetch: global loads for step i+1 issue at the top
// of the loop body (clamped address), the loadcnt wait lands after the WMMAs.
// ---------------------------------------------------------------------------
__global__ __launch_bounds__(256) void wmma_gemm(
    const _Float16* __restrict__ A, const _Float16* __restrict__ Bt,
    const float* __restrict__ bias, const float* __restrict__ res,
    float* __restrict__ Cf, _Float16* __restrict__ Ch,
    int M, int N, int K, int act) {
  __shared__ __align__(16) _Float16 As[2][64][40];
  __shared__ __align__(16) _Float16 Bs[2][128][40];
  const int tid = threadIdx.x;
  const int m0 = blockIdx.y * 64, n0 = blockIdx.x * 128;
  const int aml = tid >> 2, akseg = (tid & 3) * 8;   // A: 64 rows x 4 chunks of 8
  const int bml = tid >> 1, bkseg = (tid & 1) * 16;  // B: 128 rows x 2 chunks of 16
  const int wave = tid >> 5, lane = tid & 31, hl = lane & 15, hi = lane >> 4;
  const int sr = wave >> 1, scBase = (wave & 1) * 4;
  const _Float16* aptr = A + (size_t)imin(m0 + aml, M - 1) * K + akseg;
  const _Float16* bptr = Bt + (size_t)imin(n0 + bml, N - 1) * K + bkseg;
  v8f acc[4] = {};
  // prologue: stage k-step 0 into buffer 0
  *(v8h*)&As[0][aml][akseg]     = *(const v8h*)(aptr);
  *(v8h*)&Bs[0][bml][bkseg]     = *(const v8h*)(bptr);
  *(v8h*)&Bs[0][bml][bkseg + 8] = *(const v8h*)(bptr + 8);
  __syncthreads();
  const int nsteps = K >> 5;
  for (int i = 0; i < nsteps; ++i) {
    const int knext = imin((i + 1) * 32, K - 32);    // clamped, always valid
    v8h ra  = *(const v8h*)(aptr + knext);           // prefetch (unconditional)
    v8h rb0 = *(const v8h*)(bptr + knext);
    v8h rb1 = *(const v8h*)(bptr + knext + 8);
    const int cur = i & 1, nxt = cur ^ 1;
    const _Float16* arow = &As[cur][sr * 16 + hl][0];
    const _Float16* brow = &Bs[cur][scBase * 16 + hl][0];
    v16h a  = ld_frag(arow, hi);                     // all LDS loads first,
    v16h b0 = ld_frag(brow, hi);                     // staggered dscnt waits
    v16h b1 = ld_frag(brow + 16 * 40, hi);
    v16h b2 = ld_frag(brow + 32 * 40, hi);
    v16h b3 = ld_frag(brow + 48 * 40, hi);
    acc[0] = wmma_step(a, b0, acc[0]);
    acc[1] = wmma_step(a, b1, acc[1]);
    acc[2] = wmma_step(a, b2, acc[2]);
    acc[3] = wmma_step(a, b3, acc[3]);
    *(v8h*)&As[nxt][aml][akseg]     = ra;            // unconditional store
    *(v8h*)&Bs[nxt][bml][bkseg]     = rb0;           // (safe: buffer nxt's
    *(v8h*)&Bs[nxt][bml][bkseg + 8] = rb1;           //  readers are done)
    __syncthreads();
  }
  const int rowBase = m0 + sr * 16 + hi * 8;
#pragma unroll
  for (int j = 0; j < 4; ++j) {
    const int col = n0 + (scBase + j) * 16 + hl;
    if (col >= N) continue;
#pragma unroll
    for (int r = 0; r < 8; ++r) {
      int row = rowBase + r;
      if (row >= M) continue;
      float v = acc[j][r];
      if (bias) v += bias[col];
      if (act)  v = 0.5f * v * (1.f + erff(v * 0.70710678f));
      if (res)  v += res[(size_t)row * N + col];
      if (Cf) Cf[(size_t)row * N + col] = v;
      else    Ch[(size_t)row * N + col] = (_Float16)v;
    }
  }
}

// ---------------------------------------------------------------------------
// Attention scores: S = (Q_half · K_half^T)/8, batched over (b,h,half).
// Q/K are f16 [ROWS, 1536]; K dim = 64 -> 2 k-steps. 64x64 tile.
// ---------------------------------------------------------------------------
__global__ __launch_bounds__(256) void wmma_attn_scores(
    const _Float16* __restrict__ q, const _Float16* __restrict__ k,
    float* __restrict__ s) {
  const int z = blockIdx.z;          // (b*H + h)*2 + which
  const int which = z & 1, bh = z >> 1;
  const int h = bh % H_, b = bh / H_;
  const size_t rs = 2 * E_;
  const _Float16* qb = q + (size_t)b * NTOK * rs + h * (2 * HD_) + which * HD_;
  const _Float16* kb = k + (size_t)b * NTOK * rs + h * (2 * HD_) + which * HD_;
  float* sb = s + (size_t)z * NTOK * NTOK;
  __shared__ __align__(16) _Float16 As[64][40];
  __shared__ __align__(16) _Float16 Bs[64][40];
  const int tid = threadIdx.x;
  const int m0 = blockIdx.y * 64, n0 = blockIdx.x * 64;
  const int ml = tid >> 2, kseg = (tid & 3) * 8;
  const int wave = tid >> 5, lane = tid & 31, hl = lane & 15, hi = lane >> 4;
  const int sr = wave >> 1, sc0 = (wave & 1) * 2;
  const _Float16* qp = qb + (size_t)imin(m0 + ml, NTOK - 1) * rs + kseg;
  const _Float16* kp = kb + (size_t)imin(n0 + ml, NTOK - 1) * rs + kseg;
  v8f acc0 = {}; v8f acc1 = {};
#pragma unroll
  for (int k0 = 0; k0 < HD_; k0 += 32) {
    *(v8h*)&As[ml][kseg] = *(const v8h*)(qp + k0);
    *(v8h*)&Bs[ml][kseg] = *(const v8h*)(kp + k0);
    __syncthreads();
    v16h a  = ld_frag(&As[sr * 16 + hl][0], hi);
    v16h b0 = ld_frag(&Bs[sc0 * 16 + hl][0], hi);
    v16h b1 = ld_frag(&Bs[sc0 * 16 + 16 + hl][0], hi);
    acc0 = wmma_step(a, b0, acc0);
    acc1 = wmma_step(a, b1, acc1);
    __syncthreads();
  }
  const int colA = n0 + sc0 * 16 + hl, colB = colA + 16;
  const int rowBase = m0 + sr * 16 + hi * 8;
#pragma unroll
  for (int r = 0; r < 8; ++r) {
    int row = rowBase + r;
    if (row >= NTOK) continue;
    if (colA < NTOK) sb[(size_t)row * NTOK + colA] = 0.125f * acc0[r];
    if (colB < NTOK) sb[(size_t)row * NTOK + colB] = 0.125f * acc1[r];
  }
}

// ---------------------------------------------------------------------------
// O = attn[197,197](f32, from d_out) · V[197,128](f16), batched over (b,h).
// K=197: A zero-filled past 196 via branch-free select; B clamp is then safe.
// ---------------------------------------------------------------------------
__global__ __launch_bounds__(256) void wmma_attn_v(
    const float* __restrict__ attn, const _Float16* __restrict__ v,
    float* __restrict__ o) {
  const int z = blockIdx.z;          // b*H + h
  const int h = z % H_, b = z / H_;
  const size_t rs = 2 * E_;
  const float* ab = attn + (size_t)z * NTOK * NTOK;
  const _Float16* vb = v + (size_t)b * NTOK * rs + h * (2 * HD_);
  float* ob = o + (size_t)z * NTOK * (2 * HD_);
  __shared__ __align__(16) _Float16 As[64][40];
  __shared__ __align__(16) _Float16 Bs[64][40];
  const int tid = threadIdx.x;
  const int m0 = blockIdx.y * 64, n0 = blockIdx.x * 64;
  const int ml = tid >> 2, kseg = (tid & 3) * 8;
  const int kk = tid >> 3, nseg = (tid & 7) * 8;
  const int wave = tid >> 5, lane = tid & 31, hl = lane & 15, hi = lane >> 4;
  const int sr = wave >> 1, sc0 = (wave & 1) * 2;
  const float* arow = ab + (size_t)imin(m0 + ml, NTOK - 1) * NTOK;
  v8f acc0 = {}; v8f acc1 = {};
  for (int k0 = 0; k0 < NTOK; k0 += 32) {
#pragma unroll
    for (int j = 0; j < 8; ++j) {
      int gk = k0 + kseg + j;
      float va = arow[imin(gk, NTOK - 1)];
      As[ml][kseg + j] = (_Float16)(gk < NTOK ? va : 0.f);
    }
    v8h tv = *(const v8h*)(vb + (size_t)imin(k0 + kk, NTOK - 1) * rs + n0 + nseg);
#pragma unroll
    for (int j = 0; j < 8; ++j) Bs[nseg + j][kk] = tv[j];
    __syncthreads();
    v16h a  = ld_frag(&As[sr * 16 + hl][0], hi);
    v16h b0 = ld_frag(&Bs[sc0 * 16 + hl][0], hi);
    v16h b1 = ld_frag(&Bs[sc0 * 16 + 16 + hl][0], hi);
    acc0 = wmma_step(a, b0, acc0);
    acc1 = wmma_step(a, b1, acc1);
    __syncthreads();
  }
  const int colA = n0 + sc0 * 16 + hl, colB = colA + 16;
  const int rowBase = m0 + sr * 16 + hi * 8;
#pragma unroll
  for (int r = 0; r < 8; ++r) {
    int row = rowBase + r;
    if (row >= NTOK) continue;
    ob[(size_t)row * (2 * HD_) + colA] = acc0[r];
    ob[(size_t)row * (2 * HD_) + colB] = acc1[r];
  }
}

// ---------------------------------------------------------------------------
// Weight conversion: f32 [K,N] -> f16 transposed [N,K]; and flat f32->f16.
// ---------------------------------------------------------------------------
__global__ void convT_kernel(const float* __restrict__ in, _Float16* __restrict__ out,
                             int K, int N) {
  size_t idx = (size_t)blockIdx.x * 256 + threadIdx.x;
  if (idx >= (size_t)K * N) return;
  int k = (int)(idx / N), n = (int)(idx % N);
  out[(size_t)n * K + k] = (_Float16)in[idx];
}
__global__ void cvt_kernel(const float* __restrict__ in, _Float16* __restrict__ out,
                           size_t cnt) {
  size_t idx = (size_t)blockIdx.x * 256 + threadIdx.x;
  if (idx < cnt) out[idx] = (_Float16)in[idx];
}

// ---------------------------------------------------------------------------
// Elementwise / reduction kernels
// ---------------------------------------------------------------------------
__global__ void im2col_kernel(const float* __restrict__ x, _Float16* __restrict__ e) {
  size_t idx = (size_t)blockIdx.x * 256 + threadIdx.x;
  if (idx >= (size_t)PROWS * E_) return;
  int col = (int)(idx % E_); int row = (int)(idx / E_);
  int b = row / 196, rem = row % 196, gh = rem / 14, gw = rem % 14;
  int c = col >> 8, pq = col & 255, p = pq >> 4, qq = pq & 15;
  e[idx] = (_Float16)x[(((size_t)b * 3 + c) * 224 + gh * 16 + p) * 224 + gw * 16 + qq];
}

__global__ void addpos_kernel(const float* __restrict__ tok,
                              const float* __restrict__ cls,
                              const float* __restrict__ pos,
                              float* __restrict__ h) {
  size_t idx = (size_t)blockIdx.x * 256 + threadIdx.x;
  if (idx >= (size_t)ROWS * E_) return;
  int e = (int)(idx % E_); int row = (int)(idx / E_);
  int n = row % NTOK; int b = row / NTOK;
  float v = (n == 0) ? cls[e] : tok[((size_t)b * 196 + (n - 1)) * E_ + e];
  h[idx] = v + pos[(size_t)n * E_ + e];
}

__global__ __launch_bounds__(256) void ln_kernel(
    const float* __restrict__ in, size_t inStride, _Float16* __restrict__ out,
    const float* __restrict__ g, const float* __restrict__ bt) {
  const int row = blockIdx.x, t = threadIdx.x;
  const float* p = in + (size_t)row * inStride;
  float x0 = p[t], x1 = p[t + 256], x2 = p[t + 512];
  __shared__ float r1[256], r2[256];
  r1[t] = x0 + x1 + x2;
  r2[t] = x0 * x0 + x1 * x1 + x2 * x2;
  __syncthreads();
  for (int st = 128; st > 0; st >>= 1) {
    if (t < st) { r1[t] += r1[t + st]; r2[t] += r2[t + st]; }
    __syncthreads();
  }
  float mean = r1[0] * (1.f / 768.f);
  float var  = r2[0] * (1.f / 768.f) - mean * mean;
  float rstd = rsqrtf(var + EPS_);
  _Float16* q = out + (size_t)row * E_;
  q[t]       = (_Float16)((x0 - mean) * rstd * g[t]       + bt[t]);
  q[t + 256] = (_Float16)((x1 - mean) * rstd * g[t + 256] + bt[t + 256]);
  q[t + 512] = (_Float16)((x2 - mean) * rstd * g[t + 512] + bt[t + 512]);
}

__global__ void lambda_kernel(const float* __restrict__ lq1,
                              const float* __restrict__ lk1,
                              const float* __restrict__ lq2,
                              const float* __restrict__ lk2,
                              float* __restrict__ lam) {
  int i = blockIdx.x * blockDim.x + threadIdx.x;
  if (i < D_ * H_) {
    float s1 = 0.f, s2 = 0.f;
    for (int d = 0; d < HD_; ++d) {
      s1 += lq1[(size_t)i * HD_ + d] * lk1[(size_t)i * HD_ + d];
      s2 += lq2[(size_t)i * HD_ + d] * lk2[(size_t)i * HD_ + d];
    }
    lam[i] = expf(s1) - expf(s2) + 0.8f;
  }
}

__global__ __launch_bounds__(256) void diffsoftmax_kernel(
    const float* __restrict__ s, const float* __restrict__ lam,
    float* __restrict__ out) {
  const int rid = blockIdx.x;        // z*197 + token
  const int z = rid / NTOK, token = rid % NTOK;
  const int h = z % H_;
  const float* s1 = s + ((size_t)z * 2 + 0) * NTOK * NTOK + (size_t)token * NTOK;
  const float* s2 = s + ((size_t)z * 2 + 1) * NTOK * NTOK + (size_t)token * NTOK;
  const int t = threadIdx.x;
  float v1 = (t < NTOK) ? s1[t] : -1e30f;
  float v2 = (t < NTOK) ? s2[t] : -1e30f;
  __shared__ float red[256];
  red[t] = v1; __syncthreads();
  for (int st = 128; st > 0; st >>= 1) { if (t < st) red[t] = fmaxf(red[t], red[t + st]); __syncthreads(); }
  float m1 = red[0]; __syncthreads();
  red[t] = v2; __syncthreads();
  for (int st = 128; st > 0; st >>= 1) { if (t < st) red[t] = fmaxf(red[t], red[t + st]); __syncthreads(); }
  float m2 = red[0]; __syncthreads();
  float e1 = (t < NTOK) ? expf(v1 - m1) : 0.f;
  float e2 = (t < NTOK) ? expf(v2 - m2) : 0.f;
  red[t] = e1; __syncthreads();
  for (int st = 128; st > 0; st >>= 1) { if (t < st) red[t] += red[t + st]; __syncthreads(); }
  float d1 = red[0]; __syncthreads();
  red[t] = e2; __syncthreads();
  for (int st = 128; st > 0; st >>= 1) { if (t < st) red[t] += red[t + st]; __syncthreads(); }
  float d2 = red[0]; __syncthreads();
  if (t < NTOK)
    out[(size_t)z * NTOK * NTOK + (size_t)token * NTOK + t] =
        e1 / d1 - lam[h] * e2 / d2;
}

__global__ __launch_bounds__(256) void groupnorm_kernel(
    const float* __restrict__ o, _Float16* __restrict__ out,
    const float* __restrict__ g, const float* __restrict__ bta) {
  const int z = blockIdx.x;          // b*H + h
  const int h = z % H_, b = z / H_;
  const float* base = o + (size_t)z * NTOK * (2 * HD_);
  const int cnt = NTOK * 2 * HD_;
  __shared__ float r1[256], r2[256];
  float s = 0.f, ss = 0.f;
  for (int i = threadIdx.x; i < cnt; i += 256) { float v = base[i]; s += v; ss += v * v; }
  r1[threadIdx.x] = s; r2[threadIdx.x] = ss; __syncthreads();
  for (int st = 128; st > 0; st >>= 1) {
    if (threadIdx.x < st) { r1[threadIdx.x] += r1[threadIdx.x + st]; r2[threadIdx.x] += r2[threadIdx.x + st]; }
    __syncthreads();
  }
  float mean = r1[0] / (float)cnt;
  float var  = r2[0] / (float)cnt - mean * mean;
  float rstd = rsqrtf(var + EPS_);
  for (int i = threadIdx.x; i < cnt; i += 256) {
    int token = i >> 7, d2 = i & 127;
    float v = (base[i] - mean) * rstd * g[d2] + bta[d2];
    out[((size_t)(b * NTOK + token)) * (2 * E_) + h * (2 * HD_) + d2] = (_Float16)v;
  }
}

// ---------------------------------------------------------------------------
static inline size_t al256(size_t x) { return (x + 255) & ~(size_t)255; }

extern "C" void kernel_launch(void* const* d_in, const int* in_sizes, int n_in,
                              void* d_out, int out_size, void* d_ws, size_t ws_size,
                              hipStream_t stream) {
  const float* x      = (const float*)d_in[0];
  const float* conv_w = (const float*)d_in[1];
  const float* conv_b = (const float*)d_in[2];
  const float* cls    = (const float*)d_in[3];
  const float* pos    = (const float*)d_in[4];
  const float* Wq     = (const float*)d_in[5];
  const float* Wk     = (const float*)d_in[6];
  const float* Wv     = (const float*)d_in[7];
  const float* lq1    = (const float*)d_in[8];
  const float* lk1    = (const float*)d_in[9];
  const float* lq2    = (const float*)d_in[10];
  const float* lk2    = (const float*)d_in[11];
  const float* gn_g   = (const float*)d_in[12];
  const float* gn_b   = (const float*)d_in[13];
  const float* Wo     = (const float*)d_in[14];
  const float* ln1_g  = (const float*)d_in[15];
  const float* ln1_b  = (const float*)d_in[16];
  const float* ln2_g  = (const float*)d_in[17];
  const float* ln2_b  = (const float*)d_in[18];
  const float* w1     = (const float*)d_in[19];
  const float* b1     = (const float*)d_in[20];
  const float* w2     = (const float*)d_in[21];
  const float* b2     = (const float*)d_in[22];
  const float* lnf_g  = (const float*)d_in[23];
  const float* lnf_b  = (const float*)d_in[24];
  const float* hw     = (const float*)d_in[25];
  const float* hb     = (const float*)d_in[26];
  float* out = (float*)d_out;
  (void)in_sizes; (void)n_in; (void)out_size; (void)ws_size;

  // ---- workspace layout (byte offsets, 256B aligned) ----
  char* W = (char*)d_ws;
  size_t o = 0;
  auto allocF = [&](size_t elems) { float* p = (float*)(W + o); o = al256(o + elems * 4); return p; };
  auto allocH = [&](size_t elems) { _Float16* p = (_Float16*)(W + o); o = al256(o + elems * 2); return p; };

  float*    hbuf  = allocF((size_t)ROWS * E_);
  float*    tokb  = allocF((size_t)PROWS * E_);
  float*    sbuf  = allocF((size_t)B_ * H_ * 2 * NTOK * NTOK);
  float*    obuf  = allocF((size_t)B_ * H_ * NTOK * 2 * HD_);
  float*    lamb  = allocF(128);
  _Float16* xnbuf = allocH((size_t)ROWS * E_);
  _Float16* ebuf  = allocH((size_t)PROWS * E_);
  _Float16* qbuf  = allocH((size_t)ROWS * 2 * E_);
  _Float16* kbuf  = allocH((size_t)ROWS * 2 * E_);
  _Float16* vbuf  = allocH((size_t)ROWS * 2 * E_);
  _Float16* ob2   = allocH((size_t)ROWS * 2 * E_);
  _Float16* mbuf  = allocH((size_t)ROWS * MLP_);
  _Float16* clsb  = allocH((size_t)B_ * E_);
  _Float16* convWh= allocH((size_t)E_ * E_);
  _Float16* WqT   = allocH((size_t)D_ * 2 * E_ * E_);
  _Float16* WkT   = allocH((size_t)D_ * 2 * E_ * E_);
  _Float16* WvT   = allocH((size_t)D_ * 2 * E_ * E_);
  _Float16* WoT   = allocH((size_t)D_ * E_ * 2 * E_);
  _Float16* w1T   = allocH((size_t)D_ * MLP_ * E_);
  _Float16* w2T   = allocH((size_t)D_ * E_ * MLP_);
  _Float16* headT = allocH((size_t)NC_ * E_);

  dim3 blk(256);
  const size_t LW = (size_t)E_ * 2 * E_;   // 1,179,648
  const size_t LM = (size_t)E_ * MLP_;     // 2,359,296
  unsigned gLW = (unsigned)((LW + 255) / 256), gLM = (unsigned)((LM + 255) / 256);

  // ---- one-time weight conversion to f16 [N,K] ----
  cvt_kernel<<<dim3((unsigned)(((size_t)E_ * E_ + 255) / 256)), blk, 0, stream>>>(
      conv_w, convWh, (size_t)E_ * E_);                 // conv_w already [E(out),K]
  convT_kernel<<<dim3((unsigned)(((size_t)E_ * NC_ + 255) / 256)), blk, 0, stream>>>(
      hw, headT, E_, NC_);
  for (int l = 0; l < D_; ++l) {
    convT_kernel<<<gLW, blk, 0, stream>>>(Wq + l * LW, WqT + l * LW, E_, 2 * E_);
    convT_kernel<<<gLW, blk, 0, stream>>>(Wk + l * LW, WkT + l * LW, E_, 2 * E_);
    convT_kernel<<<gLW, blk, 0, stream>>>(Wv + l * LW, WvT + l * LW, E_, 2 * E_);
    convT_kernel<<<gLW, blk, 0, stream>>>(Wo + l * LW, WoT + l * LW, 2 * E_, E_);
    convT_kernel<<<gLM, blk, 0, stream>>>(w1 + l * LM, w1T + l * LM, E_, MLP_);
    convT_kernel<<<gLM, blk, 0, stream>>>(w2 + l * LM, w2T + l * LM, MLP_, E_);
  }

  // ---- patch embed + tokens ----
  im2col_kernel<<<dim3((unsigned)(((size_t)PROWS * E_ + 255) / 256)), blk, 0, stream>>>(x, ebuf);
  lambda_kernel<<<dim3(1), dim3(128), 0, stream>>>(lq1, lk1, lq2, lk2, lamb);
  wmma_gemm<<<dim3(E_ / 128, PROWS / 64), blk, 0, stream>>>(
      ebuf, convWh, conv_b, nullptr, tokb, nullptr, PROWS, E_, E_, 0);
  addpos_kernel<<<dim3((unsigned)(((size_t)ROWS * E_ + 255) / 256)), blk, 0, stream>>>(
      tokb, cls, pos, hbuf);

  const int gM = (ROWS + 63) / 64;             // 50
  float* attn_base = out + (size_t)B_ * NC_;   // logits occupy first 16000

  for (int l = 0; l < D_; ++l) {
    float* attn_l = attn_base + (size_t)l * B_ * H_ * NTOK * NTOK;

    ln_kernel<<<dim3(ROWS), blk, 0, stream>>>(hbuf, (size_t)E_, xnbuf,
                                              ln1_g + (size_t)l * E_, ln1_b + (size_t)l * E_);
    wmma_gemm<<<dim3(2 * E_ / 128, gM), blk, 0, stream>>>(
        xnbuf, WqT + l * LW, nullptr, nullptr, nullptr, qbuf, ROWS, 2 * E_, E_, 0);
    wmma_gemm<<<dim3(2 * E_ / 128, gM), blk, 0, stream>>>(
        xnbuf, WkT + l * LW, nullptr, nullptr, nullptr, kbuf, ROWS, 2 * E_, E_, 0);
    wmma_gemm<<<dim3(2 * E_ / 128, gM), blk, 0, stream>>>(
        xnbuf, WvT + l * LW, nullptr, nullptr, nullptr, vbuf, ROWS, 2 * E_, E_, 0);

    wmma_attn_scores<<<dim3(4, 4, B_ * H_ * 2), blk, 0, stream>>>(qbuf, kbuf, sbuf);
    diffsoftmax_kernel<<<dim3(B_ * H_ * NTOK), blk, 0, stream>>>(
        sbuf, lamb + (size_t)l * H_, attn_l);
    wmma_attn_v<<<dim3(2, 4, B_ * H_), blk, 0, stream>>>(attn_l, vbuf, obuf);
    groupnorm_kernel<<<dim3(B_ * H_), blk, 0, stream>>>(
        obuf, ob2, gn_g + (size_t)l * 2 * HD_, gn_b + (size_t)l * 2 * HD_);

    // h = h + ob2 @ Wo
    wmma_gemm<<<dim3(E_ / 128, gM), blk, 0, stream>>>(
        ob2, WoT + l * LW, nullptr, hbuf, hbuf, nullptr, ROWS, E_, 2 * E_, 0);

    ln_kernel<<<dim3(ROWS), blk, 0, stream>>>(hbuf, (size_t)E_, xnbuf,
                                              ln2_g + (size_t)l * E_, ln2_b + (size_t)l * E_);
    wmma_gemm<<<dim3(MLP_ / 128, gM), blk, 0, stream>>>(
        xnbuf, w1T + l * LM, b1 + (size_t)l * MLP_, nullptr, nullptr, mbuf,
        ROWS, MLP_, E_, 1);  // fused GELU, f16 out
    wmma_gemm<<<dim3(E_ / 128, gM), blk, 0, stream>>>(
        mbuf, w2T + l * LM, b2 + (size_t)l * E_, hbuf, hbuf, nullptr,
        ROWS, E_, MLP_, 0);  // residual add
  }

  // ---- final LN on the 16 cls rows, head GEMM into logits ----
  ln_kernel<<<dim3(B_), blk, 0, stream>>>(hbuf, (size_t)NTOK * E_, clsb, lnf_g, lnf_b);
  wmma_gemm<<<dim3((NC_ + 127) / 128, 1), blk, 0, stream>>>(
      clsb, headT, hb, nullptr, out, nullptr, B_, NC_, E_, 0);
}
